// DecoderDefAttnBlock_75170517615231
// MI455X (gfx1250) — compile-verified, matrix operand order checked by reading, not run on hardware
//
#include <hip/hip_runtime.h>

typedef __attribute__((ext_vector_type(16))) _Float16 v16h;
typedef __attribute__((ext_vector_type(8)))  float    v8f;
typedef __attribute__((ext_vector_type(2)))  _Float16 h2;
typedef __attribute__((ext_vector_type(4)))  _Float16 h4;
typedef __attribute__((ext_vector_type(8)))  _Float16 h8;

// ---------------- problem constants ----------------
static constexpr int BATCH = 2;
static constexpr int C     = 256;
static constexpr int NH    = 8;
static constexpr int NLVL  = 3;
static constexpr int NPT   = 4;
static constexpr int NOFF  = NH * NLVL * NPT * 3;  // 288
static constexpr int NATT  = NH * NLVL * NPT;      // 96
static constexpr int DFFN  = 1024;
static constexpr int S0 = 12 * 24 * 24;            // 6912
static constexpr int S1 = 6 * 12 * 12;             // 864
static constexpr int S2 = 3 * 6 * 6;               // 108
static constexpr int S  = S0 + S1 + S2;            // 7884
static constexpr int M  = BATCH * S;               // 15768

// ---------------- pack / unpack ----------------
__global__ void pack_kernel(const float* __restrict__ f0, const float* __restrict__ f1,
                            const float* __restrict__ f2, const float* __restrict__ p0,
                            const float* __restrict__ p1, const float* __restrict__ p2,
                            const float* __restrict__ le, float* __restrict__ src,
                            _Float16* __restrict__ srch, float* __restrict__ pos) {
  int i = blockIdx.x * blockDim.x + threadIdx.x;
  if (i >= M * C) return;
  int c = i % C;
  int s = (i / C) % S;
  int b = i / (C * S);
  int l, sp, n;
  const float *f, *p;
  if (s < S0)            { l = 0; sp = s;           n = S0; f = f0; p = p0; }
  else if (s < S0 + S1)  { l = 1; sp = s - S0;      n = S1; f = f1; p = p1; }
  else                   { l = 2; sp = s - S0 - S1; n = S2; f = f2; p = p2; }
  size_t fi = ((size_t)b * C + c) * n + sp;
  float v = f[fi];
  src[i]  = v;
  srch[i] = (_Float16)v;
  pos[i]  = p[fi] + le[l * C + c];
}

__global__ void unpack_kernel(const float* __restrict__ src, float* __restrict__ out) {
  int i = blockIdx.x * blockDim.x + threadIdx.x;
  if (i >= M * C) return;
  constexpr int C0 = BATCH * C * S0, C1 = BATCH * C * S1;
  int t, n, st;
  if (i < C0)           { t = i;           n = S0; st = 0; }
  else if (i < C0 + C1) { t = i - C0;      n = S1; st = S0; }
  else                  { t = i - C0 - C1; n = S2; st = S0 + S1; }
  int sp = t % n;
  int c  = (t / n) % C;
  int b  = t / (n * C);
  out[i] = src[((size_t)(b * S + st + sp)) * C + c];
}

// ---------------- elementwise ----------------
__global__ void add_h_kernel(const float* __restrict__ a, const float* __restrict__ b,
                             _Float16* __restrict__ o, int n) {
  int i = blockIdx.x * blockDim.x + threadIdx.x;
  if (i < n) o[i] = (_Float16)(a[i] + b[i]);
}

// weight convert + transpose: Wt[n*K + k] = (f16) W[k*N + n]
__global__ void wconv_kernel(const float* __restrict__ W, _Float16* __restrict__ Wt,
                             int K, int N) {
  int i = blockIdx.x * blockDim.x + threadIdx.x;
  if (i >= N * K) return;
  int n = i / K, k = i % K;
  Wt[i] = (_Float16)W[(size_t)k * N + n];
}

// ---------------- WMMA GEMM ----------------
// Cout[M,N] = A[M,K](f16) @ Wt[N,K](f16, pre-transposed) + bias, opt ReLU, f32 or f16 out.
// Block tile 128x32: 8 waves, each wave = one 16-row strip x two 16-col tiles
// (A fragment reused across 2 WMMAs per K-step).
template <bool RELU, bool HALFOUT>
__global__ void gemm_wmma_kernel(const _Float16* __restrict__ A,
                                 const _Float16* __restrict__ Wt,
                                 const float* __restrict__ bias, void* __restrict__ outp,
                                 int Mm, int K, int N) {
  constexpr int LDA = 40, LDB = 40;  // halves; 80B row stride keeps b128 stores 16B-aligned
  __shared__ _Float16 As[128 * LDA];
  __shared__ _Float16 Bs[32 * LDB];
  const int t    = threadIdx.x;
  const int wid  = t >> 5, lane = t & 31;
  const int m0   = blockIdx.y * 128;
  const int n0   = blockIdx.x * 32;
  const int hwv  = lane >> 4, r16 = lane & 15;
  v8f acc0 = {0.f, 0.f, 0.f, 0.f, 0.f, 0.f, 0.f, 0.f};
  v8f acc1 = {0.f, 0.f, 0.f, 0.f, 0.f, 0.f, 0.f, 0.f};

  const int ar = t >> 1, akq = (t & 1) * 16;  // A stage: row ar, 16 halves from akq
  const int bn = t >> 3, bkq = (t & 7) * 4;   // B stage: row n, 4 halves from bkq

  for (int k0 = 0; k0 < K; k0 += 32) {
    __syncthreads();
    // stage A tile (f16, b128 copies)
    if (m0 + ar < Mm) {
      const h8* ap = reinterpret_cast<const h8*>(A + (size_t)(m0 + ar) * K + k0 + akq);
      h8 x = ap[0], y = ap[1];
      *reinterpret_cast<h8*>(&As[ar * LDA + akq])     = x;
      *reinterpret_cast<h8*>(&As[ar * LDA + akq + 8]) = y;
      if (k0 + 32 < K)  // prefetch next K chunk (speculative, gfx1250 global_prefetch)
        __builtin_prefetch(A + (size_t)(m0 + ar) * K + k0 + 32 + akq, 0, 3);
    } else {
      h8 z = {};
      *reinterpret_cast<h8*>(&As[ar * LDA + akq])     = z;
      *reinterpret_cast<h8*>(&As[ar * LDA + akq + 8]) = z;
    }
    // stage B tile: Bs[n][k] directly from pre-transposed Wt[N][K]
    {
      h4 x = *reinterpret_cast<const h4*>(Wt + (size_t)(n0 + bn) * K + k0 + bkq);
      *reinterpret_cast<h4*>(&Bs[bn * LDB + bkq]) = x;
    }
    __syncthreads();

    // fragments: ISA 16-bit layout, K = (e/8)*16 + halfwave*8 + e%8
    v16h a, b0, b1;
    const int am = wid * 16 + r16;
    #pragma unroll
    for (int j = 0; j < 8; ++j) {
      const int kb = ((j >> 2) << 4) + (hwv << 3) + ((j & 3) << 1);
      h2 av = *reinterpret_cast<const h2*>(&As[am * LDA + kb]);
      h2 bv0 = *reinterpret_cast<const h2*>(&Bs[r16 * LDB + kb]);
      h2 bv1 = *reinterpret_cast<const h2*>(&Bs[(16 + r16) * LDB + kb]);
      a[2 * j]  = av[0];  a[2 * j + 1]  = av[1];
      b0[2 * j] = bv0[0]; b0[2 * j + 1] = bv0[1];
      b1[2 * j] = bv1[0]; b1[2 * j + 1] = bv1[1];
    }
    acc0 = __builtin_amdgcn_wmma_f32_16x16x32_f16(false, a, false, b0, (short)0, acc0,
                                                  false, false);
    acc1 = __builtin_amdgcn_wmma_f32_16x16x32_f16(false, a, false, b1, (short)0, acc1,
                                                  false, false);
  }

  const int col0 = n0 + r16, col1 = n0 + 16 + r16;
  const float bv0 = bias[col0], bv1 = bias[col1];
  #pragma unroll
  for (int r = 0; r < 8; ++r) {
    int row = m0 + wid * 16 + hwv * 8 + r;
    if (row < Mm) {
      float v0 = acc0[r] + bv0;
      float v1 = acc1[r] + bv1;
      if (RELU) { v0 = v0 > 0.f ? v0 : 0.f; v1 = v1 > 0.f ? v1 : 0.f; }
      if (HALFOUT) {
        _Float16* o = (_Float16*)outp;
        o[(size_t)row * N + col0] = (_Float16)v0;
        o[(size_t)row * N + col1] = (_Float16)v1;
      } else {
        float* o = (float*)outp;
        o[(size_t)row * N + col0] = v0;
        o[(size_t)row * N + col1] = v1;
      }
    }
  }
}

// ---------------- softmax over 12 (per token, head) ----------------
__global__ void softmax_kernel(float* __restrict__ attn) {
  int i = blockIdx.x * blockDim.x + threadIdx.x;
  if (i >= M * NH) return;
  float* p = attn + (size_t)(i / NH) * NATT + (i % NH) * (NLVL * NPT);
  float v[12], mx = -1e30f;
  #pragma unroll
  for (int j = 0; j < 12; ++j) { v[j] = p[j]; mx = v[j] > mx ? v[j] : mx; }
  float sum = 0.f;
  #pragma unroll
  for (int j = 0; j < 12; ++j) { v[j] = expf(v[j] - mx); sum += v[j]; }
  float inv = 1.f / sum;
  #pragma unroll
  for (int j = 0; j < 12; ++j) p[j] = v[j] * inv;
}

// ---------------- layernorm(x + res)*g + b, one wave per token; f32 + f16 outputs ----------------
__global__ void ln_kernel(const float* __restrict__ x, const float* __restrict__ res,
                          const float* __restrict__ g, const float* __restrict__ beta,
                          float* __restrict__ out, _Float16* __restrict__ outh) {
  int w    = (blockIdx.x * blockDim.x + threadIdx.x) >> 5;
  int lane = threadIdx.x & 31;
  if (w >= M) return;
  const float* xp = x + (size_t)w * C;
  const float* rp = res + (size_t)w * C;
  float v[8], sum = 0.f, sq = 0.f;
  #pragma unroll
  for (int j = 0; j < 8; ++j) {
    int c = lane + 32 * j;
    float t = xp[c] + rp[c];
    v[j] = t; sum += t; sq += t * t;
  }
  #pragma unroll
  for (int off = 16; off > 0; off >>= 1) {
    sum += __shfl_xor(sum, off, 32);
    sq  += __shfl_xor(sq,  off, 32);
  }
  float mean = sum * (1.f / C);
  float var  = sq * (1.f / C) - mean * mean;
  float rstd = rsqrtf(var + 1e-5f);
  float*    op = out  + (size_t)w * C;
  _Float16* oh = outh + (size_t)w * C;
  #pragma unroll
  for (int j = 0; j < 8; ++j) {
    int c = lane + 32 * j;
    float o = (v[j] - mean) * rstd * g[c] + beta[c];
    op[c] = o;
    oh[c] = (_Float16)o;
  }
}

// ---------------- deformable trilinear sampling: one wave per (b, q, head) ----------------
__device__ __forceinline__ void ref_point(int q, float& rx, float& ry, float& rz) {
  if (q < S0) {
    int z = q / 576, r = q % 576, y = r / 24, x = r % 24;
    rx = (x + 0.5f) / 24.f; ry = (y + 0.5f) / 24.f; rz = (z + 0.5f) / 12.f;
  } else if (q < S0 + S1) {
    int t = q - S0, z = t / 144, r = t % 144, y = r / 12, x = r % 12;
    rx = (x + 0.5f) / 12.f; ry = (y + 0.5f) / 12.f; rz = (z + 0.5f) / 6.f;
  } else {
    int t = q - S0 - S1, z = t / 36, r = t % 36, y = r / 6, x = r % 6;
    rx = (x + 0.5f) / 6.f; ry = (y + 0.5f) / 6.f; rz = (z + 0.5f) / 3.f;
  }
}

__global__ void deform_kernel(const float* __restrict__ value, const float* __restrict__ offs,
                              const float* __restrict__ attn, _Float16* __restrict__ out) {
  int gw   = (blockIdx.x * blockDim.x + threadIdx.x) >> 5;
  int lane = threadIdx.x & 31;
  if (gw >= M * NH) return;
  const int head = gw % NH;
  const int q    = (gw / NH) % S;
  const int b    = gw / (NH * S);
  const int LD_[3] = {12, 6, 3}, LH_[3] = {24, 12, 6}, LW_[3] = {24, 12, 6};
  const int LST[3] = {0, S0, S0 + S1};
  float rx, ry, rz;
  ref_point(q, rx, ry, rz);
  const size_t tok  = (size_t)b * S + q;
  const float* offp = offs + tok * NOFF + head * (NLVL * NPT * 3);
  const float* attp = attn + tok * NATT + head * (NLVL * NPT);
  float acc = 0.f;
  for (int l = 0; l < NLVL; ++l) {
    const int D_ = LD_[l], H_ = LH_[l], W_ = LW_[l];
    const float* vbase = value + ((size_t)b * S + LST[l]) * C + head * 32 + lane;
    for (int p = 0; p < NPT; ++p) {
      const float* o = offp + (l * NPT + p) * 3;
      float aw = attp[l * NPT + p];
      float xs = rx * W_ + o[0] - 0.5f;
      float ys = ry * H_ + o[1] - 0.5f;
      float zs = rz * D_ + o[2] - 0.5f;
      float xf = floorf(xs), yf = floorf(ys), zf = floorf(zs);
      float fx = xs - xf, fy = ys - yf, fz = zs - zf;
      int x0 = (int)xf, y0 = (int)yf, z0 = (int)zf;
      float sres = 0.f;
      #pragma unroll
      for (int dz = 0; dz < 2; ++dz) {
        int zi = z0 + dz;
        if (zi < 0 || zi >= D_) continue;
        float wz = dz ? fz : 1.f - fz;
        #pragma unroll
        for (int dy = 0; dy < 2; ++dy) {
          int yi = y0 + dy;
          if (yi < 0 || yi >= H_) continue;
          float wy = dy ? fy : 1.f - fy;
          #pragma unroll
          for (int dx = 0; dx < 2; ++dx) {
            int xi = x0 + dx;
            if (xi < 0 || xi >= W_) continue;
            float wx = dx ? fx : 1.f - fx;
            sres += wz * wy * wx * vbase[(size_t)((zi * H_ + yi) * W_ + xi) * C];
          }
        }
      }
      acc += aw * sres;
    }
  }
  out[tok * C + head * 32 + lane] = (_Float16)acc;
}

// ---------------- host driver ----------------
static void launch_gemm(const _Float16* A, const _Float16* Wt, const float* bias, void* out,
                        int Mm, int K, int N, bool relu_half, hipStream_t s) {
  dim3 g(N / 32, (Mm + 127) / 128);
  if (relu_half)
    gemm_wmma_kernel<true, true><<<g, 256, 0, s>>>(A, Wt, bias, out, Mm, K, N);
  else
    gemm_wmma_kernel<false, false><<<g, 256, 0, s>>>(A, Wt, bias, out, Mm, K, N);
}

static void launch_wconv(const float* W, _Float16* Wt, int K, int N, hipStream_t s) {
  int n = K * N;
  wconv_kernel<<<(n + 255) / 256, 256, 0, s>>>(W, Wt, K, N);
}

extern "C" void kernel_launch(void* const* d_in, const int* in_sizes, int n_in,
                              void* d_out, int out_size, void* d_ws, size_t ws_size,
                              hipStream_t stream) {
  (void)in_sizes; (void)n_in; (void)out_size; (void)ws_size;
  const float* f0 = (const float*)d_in[0];
  const float* p0 = (const float*)d_in[1];
  const float* f1 = (const float*)d_in[2];
  const float* p1 = (const float*)d_in[3];
  const float* f2 = (const float*)d_in[4];
  const float* p2 = (const float*)d_in[5];
  const float* le   = (const float*)d_in[6];
  const float* Wo   = (const float*)d_in[7];
  const float* bo   = (const float*)d_in[8];
  const float* Wa   = (const float*)d_in[9];
  const float* ba   = (const float*)d_in[10];
  const float* Wv   = (const float*)d_in[11];
  const float* bv   = (const float*)d_in[12];
  const float* Wout = (const float*)d_in[13];
  const float* bout = (const float*)d_in[14];
  const float* ln1g = (const float*)d_in[15];
  const float* ln1b = (const float*)d_in[16];
  const float* W1   = (const float*)d_in[17];
  const float* b1   = (const float*)d_in[18];
  const float* W2   = (const float*)d_in[19];
  const float* b2   = (const float*)d_in[20];
  const float* ln2g = (const float*)d_in[21];
  const float* ln2b = (const float*)d_in[22];

  const size_t MC = (size_t)M * C;
  char*  base = (char*)d_ws;
  size_t off  = 0;
  auto alloc = [&](size_t bytes) -> void* {
    void* p = base + off;
    off += (bytes + 255) & ~(size_t)255;
    return p;
  };
  float* src  = (float*)alloc(MC * 4);
  float* pos  = (float*)alloc(MC * 4);
  float* val  = (float*)alloc(MC * 4);
  float* src2 = (float*)alloc(MC * 4);
  float* srcB = (float*)alloc(MC * 4);
  float* offs = (float*)alloc((size_t)M * NOFF * 4);
  float* attn = (float*)alloc((size_t)M * NATT * 4);
  _Float16* srch  = (_Float16*)alloc(MC * 2);
  _Float16* qh    = (_Float16*)alloc(MC * 2);
  _Float16* samph = (_Float16*)alloc(MC * 2);
  _Float16* srcBh = (_Float16*)alloc(MC * 2);
  _Float16* hbuf  = (_Float16*)alloc((size_t)M * DFFN * 2);
  _Float16* WvT = (_Float16*)alloc((size_t)C * C * 2);
  _Float16* WoT = (_Float16*)alloc((size_t)C * NOFF * 2);
  _Float16* WaT = (_Float16*)alloc((size_t)C * NATT * 2);
  _Float16* WqT = (_Float16*)alloc((size_t)C * C * 2);
  _Float16* W1T = (_Float16*)alloc((size_t)C * DFFN * 2);
  _Float16* W2T = (_Float16*)alloc((size_t)DFFN * C * 2);

  const int EB  = 256;
  const int nMC = (int)MC;
  const int gMC = (nMC + EB - 1) / EB;

  pack_kernel<<<gMC, EB, 0, stream>>>(f0, f1, f2, p0, p1, p2, le, src, srch, pos);

  for (int i = 0; i < 2; ++i) {
    const float* Wo_l = Wo + (size_t)i * C * NOFF;
    const float* bo_l = bo + (size_t)i * NOFF;
    const float* Wa_l = Wa + (size_t)i * C * NATT;
    const float* ba_l = ba + (size_t)i * NATT;
    const float* Wv_l = Wv + (size_t)i * C * C;
    const float* bv_l = bv + (size_t)i * C;
    const float* Wq_l = Wout + (size_t)i * C * C;
    const float* bq_l = bout + (size_t)i * C;
    const float* g1   = ln1g + (size_t)i * C;
    const float* be1  = ln1b + (size_t)i * C;
    const float* W1_l = W1 + (size_t)i * C * DFFN;
    const float* b1_l = b1 + (size_t)i * DFFN;
    const float* W2_l = W2 + (size_t)i * DFFN * C;
    const float* b2_l = b2 + (size_t)i * C;
    const float* g2   = ln2g + (size_t)i * C;
    const float* be2  = ln2b + (size_t)i * C;

    // convert + transpose this layer's weights to f16 [N][K]
    launch_wconv(Wv_l, WvT, C, C, stream);
    launch_wconv(Wo_l, WoT, C, NOFF, stream);
    launch_wconv(Wa_l, WaT, C, NATT, stream);
    launch_wconv(Wq_l, WqT, C, C, stream);
    launch_wconv(W1_l, W1T, C, DFFN, stream);
    launch_wconv(W2_l, W2T, DFFN, C, stream);

    // q = f16(src + pos)
    add_h_kernel<<<gMC, EB, 0, stream>>>(src, pos, qh, nMC);
    // value = src @ Wv + bv   (f32 out, consumed by gather)
    launch_gemm(srch, WvT, bv_l, val, M, C, C, false, stream);
    // offsets & attention logits from q
    launch_gemm(qh, WoT, bo_l, offs, M, C, NOFF, false, stream);
    launch_gemm(qh, WaT, ba_l, attn, M, C, NATT, false, stream);
    softmax_kernel<<<(M * NH + EB - 1) / EB, EB, 0, stream>>>(attn);
    // deformable sampling -> f16
    deform_kernel<<<(M * NH * 32 + EB - 1) / EB, EB, 0, stream>>>(val, offs, attn, samph);
    // output projection
    launch_gemm(samph, WqT, bq_l, src2, M, C, C, false, stream);
    // src = LN(src + src2)
    ln_kernel<<<(M * 32 + EB - 1) / EB, EB, 0, stream>>>(src, src2, g1, be1, srcB, srcBh);
    // FFN: h = relu(srcB @ W1 + b1) in f16; ffo = h @ W2 + b2 in f32
    launch_gemm(srcBh, W1T, b1_l, hbuf, M, C, DFFN, true, stream);
    launch_gemm(hbuf, W2T, b2_l, src2, M, DFFN, C, false, stream);
    ln_kernel<<<(M * 32 + EB - 1) / EB, EB, 0, stream>>>(srcB, src2, g2, be2, src, srch);
  }

  unpack_kernel<<<gMC, EB, 0, stream>>>(src, (float*)d_out);
}
